// Parallel_Multiheaded_Self_Attention_56410100466003
// MI455X (gfx1250) — compile-verified, matrix operand order checked by reading, not run on hardware
//
#include <hip/hip_runtime.h>
#include <hip/hip_bf16.h>

typedef __bf16 bf16_t;
typedef __attribute__((ext_vector_type(16))) __bf16 v16bf;
typedef __attribute__((ext_vector_type(8)))  __bf16 v8bf;
typedef __attribute__((ext_vector_type(8)))  float   v8f;
typedef __attribute__((ext_vector_type(4)))  int     v4i;

#define DM   2048   // d_model
#define SL   2048   // seq len
#define NB   4      // batch
#define NH   16     // heads
#define DH   128    // head dim
#define BSM  (NB*SL) // 8192 total tokens

// ---------------------------------------------------------------------------
// CDNA5 async global->LDS copy (GLOBAL_LOAD_ASYNC_TO_LDS_B128, ASYNCcnt).
// clang-22 signature (from diagnostic): (v4i addrspace(1)*, v4i addrspace(3)*,
// imm offset, imm cpol). Guarded by __has_builtin with a plain-store fallback.
// ---------------------------------------------------------------------------
#if __has_builtin(__builtin_amdgcn_global_load_async_to_lds_b128)
#define HAS_ASYNC_LDS 1
#else
#define HAS_ASYNC_LDS 0
#endif

typedef __attribute__((address_space(1))) v4i* gv4i_p;
typedef __attribute__((address_space(3))) v4i* lv4i_p;

static __device__ __forceinline__ void async_copy16(bf16_t* l, const bf16_t* g) {
#if HAS_ASYNC_LDS
  __builtin_amdgcn_global_load_async_to_lds_b128((gv4i_p)g, (lv4i_p)l,
                                                 /*offset=*/0, /*cpol=*/0);
#else
  *(v8bf*)l = *(const v8bf*)g;
#endif
}

static __device__ __forceinline__ void async_wait() {
#if HAS_ASYNC_LDS
#if __has_builtin(__builtin_amdgcn_s_wait_asynccnt)
  __builtin_amdgcn_s_wait_asynccnt(0);
#else
  asm volatile("s_wait_asynccnt 0x0" ::: "memory");
#endif
#endif
}

// ---------------------------------------------------------------------------
// WMMA fragment loader: both A ([M][K], K-contiguous rows) and B ([N][K])
// fragments for v_wmma_*_16x16x32_bf16 load as two 16B chunks per lane:
//   elems 0..7  = K [8*lh .. 8*lh+7],  elems 8..15 = K [16+8*lh .. 16+8*lh+7]
// ---------------------------------------------------------------------------
static __device__ __forceinline__ v16bf load_frag(const bf16_t* base, int lh) {
  union { v16bf v; v8bf h[2]; } u;
  u.h[0] = *(const v8bf*)(base + 8 * lh);
  u.h[1] = *(const v8bf*)(base + 16 + 8 * lh);
  return u.v;
}

static __device__ __forceinline__ v8f wmma_bf16(v16bf a, v16bf b, v8f c) {
  // (neg_a, A, neg_b, B, c_mod, C, reuse_a, reuse_b)
  return __builtin_amdgcn_wmma_f32_16x16x32_bf16(false, a, false, b, (short)0, c,
                                                 false, false);
}

// ---------------------------------------------------------------------------
// f32 -> bf16 conversion (vectorized by 4)
// ---------------------------------------------------------------------------
__global__ __launch_bounds__(256) void cvt_kernel(const float* __restrict__ in,
                                                  bf16_t* __restrict__ out,
                                                  int n4) {
  int i = blockIdx.x * 256 + threadIdx.x;
  if (i < n4) {
    float4 f = ((const float4*)in)[i];
    out[4 * i + 0] = (bf16_t)f.x;
    out[4 * i + 1] = (bf16_t)f.y;
    out[4 * i + 2] = (bf16_t)f.z;
    out[4 * i + 3] = (bf16_t)f.w;
  }
}

// ---------------------------------------------------------------------------
// Double-buffered GEMM mainloop: C[128x128] = A[128xK] * B[128xK]^T (K=2048).
// A: [M][K] row-major bf16, Bm: [N][K] row-major bf16.
// 256 threads = 8 waves in 4(M) x 2(N); each wave owns 32x64.
// As/Bs are [2][128*64] LDS buffers; tile t+1 streams in asynchronously
// while tile t is consumed by WMMA.
// ---------------------------------------------------------------------------
static __device__ __forceinline__ void stage_gemm_tile(
    const bf16_t* __restrict__ A, const bf16_t* __restrict__ Bm,
    bf16_t* As, bf16_t* Bs, int bM, int bN, int k0, int tid) {
  const int c = (tid & 7) * 8;
#pragma unroll
  for (int p = 0; p < 4; p++) {
    const int r = (tid >> 3) + p * 32;
    async_copy16(As + r * 64 + c, A + (size_t)(bM * 128 + r) * DM + k0 + c);
    async_copy16(Bs + r * 64 + c, Bm + (size_t)(bN * 128 + r) * DM + k0 + c);
  }
}

static __device__ __forceinline__ void gemm_tile_core(
    const bf16_t* __restrict__ A, const bf16_t* __restrict__ Bm,
    bf16_t* As, bf16_t* Bs,  // each 2*128*64 bf16
    int bM, int bN, int tid, v8f acc[2][4]) {
  const int lane = tid & 31, lr = lane & 15, lh = lane >> 4;
  const int wid = tid >> 5, wm = wid & 3, wn = wid >> 2;

#pragma unroll
  for (int i = 0; i < 2; i++)
#pragma unroll
    for (int j = 0; j < 4; j++) {
      v8f z = {};
      acc[i][j] = z;
    }

  stage_gemm_tile(A, Bm, As, Bs, bM, bN, 0, tid);
  async_wait();
  __syncthreads();

  for (int k0 = 0; k0 < DM; k0 += 64) {
    const int cur = (k0 >> 6) & 1;
    const bf16_t* Ac = As + cur * (128 * 64);
    const bf16_t* Bc = Bs + cur * (128 * 64);
    if (k0 + 64 < DM)
      stage_gemm_tile(A, Bm, As + (1 - cur) * (128 * 64),
                      Bs + (1 - cur) * (128 * 64), bM, bN, k0 + 64, tid);

#pragma unroll
    for (int ks = 0; ks < 64; ks += 32) {
      v16bf af[2], bfr[4];
#pragma unroll
      for (int i = 0; i < 2; i++)
        af[i] = load_frag(Ac + (wm * 32 + i * 16 + lr) * 64 + ks, lh);
#pragma unroll
      for (int j = 0; j < 4; j++)
        bfr[j] = load_frag(Bc + (wn * 64 + j * 16 + lr) * 64 + ks, lh);
#pragma unroll
      for (int i = 0; i < 2; i++)
#pragma unroll
        for (int j = 0; j < 4; j++)
          acc[i][j] = wmma_bf16(af[i], bfr[j], acc[i][j]);
    }
    async_wait();
    __syncthreads();
  }
}

// ---------------------------------------------------------------------------
// QKV projection: qkv[k] = X * W[k]^T.
//  k=0 -> Q in [b,h,s,dh]; k=1 -> K in [b,h,s,dh]; k=2 -> V^T in [b,h,dh,s]
// ---------------------------------------------------------------------------
__global__ __launch_bounds__(256) void qkv_gemm_kernel(
    const bf16_t* __restrict__ X, const bf16_t* __restrict__ W3,
    bf16_t* __restrict__ Qo, bf16_t* __restrict__ Ko, bf16_t* __restrict__ VTo) {
  __shared__ bf16_t As[2 * 128 * 64];
  __shared__ bf16_t Bs[2 * 128 * 64];
  const int tid = threadIdx.x;
  const int km = blockIdx.z;
  const bf16_t* Bm = W3 + (size_t)km * DM * DM;
  v8f acc[2][4];
  gemm_tile_core(X, Bm, As, Bs, blockIdx.y, blockIdx.x, tid, acc);

  const int wid = tid >> 5, lane = tid & 31, lr = lane & 15, lh = lane >> 4;
  const int wm = wid & 3, wn = wid >> 2;
  bf16_t* QK = (km == 0) ? Qo : Ko;
#pragma unroll
  for (int i = 0; i < 2; i++)
#pragma unroll
    for (int j = 0; j < 4; j++)
#pragma unroll
      for (int v = 0; v < 8; v++) {
        const int Mg = blockIdx.y * 128 + wm * 32 + i * 16 + v + 8 * lh;
        const int Ng = blockIdx.x * 128 + wn * 64 + j * 16 + lr;
        const int b = Mg >> 11, s = Mg & (SL - 1);
        const int h = Ng >> 7, dh = Ng & (DH - 1);
        const float val = acc[i][j][v];
        if (km < 2)
          QK[(((size_t)(b * NH + h)) * SL + s) * DH + dh] = (bf16_t)val;
        else
          VTo[(((size_t)(b * NH + h)) * DH + dh) * SL + s] = (bf16_t)val;
      }
}

// ---------------------------------------------------------------------------
// Output projection: Out = AO * Wo^T, f32 result (row-major [t][d])
// ---------------------------------------------------------------------------
__global__ __launch_bounds__(256) void out_gemm_kernel(
    const bf16_t* __restrict__ Ain, const bf16_t* __restrict__ Wo,
    float* __restrict__ Out) {
  __shared__ bf16_t As[2 * 128 * 64];
  __shared__ bf16_t Bs[2 * 128 * 64];
  const int tid = threadIdx.x;
  v8f acc[2][4];
  gemm_tile_core(Ain, Wo, As, Bs, blockIdx.y, blockIdx.x, tid, acc);

  const int wid = tid >> 5, lane = tid & 31, lr = lane & 15, lh = lane >> 4;
  const int wm = wid & 3, wn = wid >> 2;
#pragma unroll
  for (int i = 0; i < 2; i++)
#pragma unroll
    for (int j = 0; j < 4; j++)
#pragma unroll
      for (int v = 0; v < 8; v++) {
        const int Mg = blockIdx.y * 128 + wm * 32 + i * 16 + v + 8 * lh;
        const int Ng = blockIdx.x * 128 + wn * 64 + j * 16 + lr;
        Out[(size_t)Mg * DM + Ng] = acc[i][j][v];
      }
}

// ---------------------------------------------------------------------------
// RoPE applied in place to Q and K ([b,h,s,dh] bf16).
// ---------------------------------------------------------------------------
__global__ __launch_bounds__(256) void rope_kernel(bf16_t* __restrict__ Qb,
                                                   bf16_t* __restrict__ Kb) {
  const unsigned idx = blockIdx.x * 256u + threadIdx.x;
  const int i = idx & 63;
  const int s = (idx >> 6) & (SL - 1);
  const int bh = idx >> 17;
  // inv_freq = 10000^(-2i/128) = exp(-2i/128 * ln(10000))
  const float ang =
      (float)s * __expf(-(float)(2 * i) * (9.210340371976184f / 128.0f));
  float sn, cs;
  __sincosf(ang, &sn, &cs);
  const size_t base = ((size_t)bh * SL + s) * DH + 2 * i;
  {
    float x0 = (float)Qb[base], x1 = (float)Qb[base + 1];
    Qb[base] = (bf16_t)(cs * x0 - sn * x1);
    Qb[base + 1] = (bf16_t)(sn * x0 + cs * x1);
  }
  {
    float x0 = (float)Kb[base], x1 = (float)Kb[base + 1];
    Kb[base] = (bf16_t)(cs * x0 - sn * x1);
    Kb[base + 1] = (bf16_t)(sn * x0 + cs * x1);
  }
}

// ---------------------------------------------------------------------------
// Causal flash attention. One block = one (b,h) and 128 query rows.
// 8 waves x 16 query rows each; 32-key tiles (double-buffered async into
// LDS); online softmax in f32; Q*K^T and P*V on v_wmma_f32_16x16x32_bf16.
// ---------------------------------------------------------------------------
static __device__ __forceinline__ void stage_kv(
    const bf16_t* __restrict__ Kv, const bf16_t* __restrict__ VT,
    bf16_t* Ksb, bf16_t* Vsb, size_t headBase, size_t headBaseT, int kv0,
    int tid) {
  const int r = tid >> 3;           // 0..31 key rows
  const int c = (tid & 7) * 16;     // dh columns
  async_copy16(Ksb + r * 128 + c, Kv + (headBase + kv0 + r) * DH + c);
  async_copy16(Ksb + r * 128 + c + 8, Kv + (headBase + kv0 + r) * DH + c + 8);
  const int rv = tid >> 1;          // 0..127 dh rows
  const int cv = (tid & 1) * 16;    // key columns
  async_copy16(Vsb + rv * 32 + cv, VT + (headBaseT + rv) * SL + kv0 + cv);
  async_copy16(Vsb + rv * 32 + cv + 8, VT + (headBaseT + rv) * SL + kv0 + cv + 8);
}

__global__ __launch_bounds__(256) void flash_kernel(
    const bf16_t* __restrict__ Q, const bf16_t* __restrict__ Kv,
    const bf16_t* __restrict__ VT, bf16_t* __restrict__ AO) {
  __shared__ bf16_t Ks[2 * 32 * 128];   // [buf][key][dh]
  __shared__ bf16_t Vs[2 * 128 * 32];   // [buf][dh][key]
  __shared__ bf16_t Ps[8 * 16 * 32];    // per-wave P staging (C-frag -> A-frag)

  const int tid = threadIdx.x;
  const int wid = tid >> 5, lane = tid & 31, lr = lane & 15, lh = lane >> 4;
  const int qt = blockIdx.x, h = blockIdx.y, b = blockIdx.z;
  const int q0 = qt * 128;
  const size_t headBase = (size_t)(b * NH + h) * SL;   // rows of Q/K
  const size_t headBaseT = (size_t)(b * NH + h) * DH;  // rows of V^T

  // resident Q fragments: 16 rows x 128 dh = 4 k-fragments
  const bf16_t* qptr = Q + (headBase + (q0 + wid * 16 + lr)) * DH;
  v16bf qf[4];
#pragma unroll
  for (int kk = 0; kk < 4; kk++) qf[kk] = load_frag(qptr + kk * 32, lh);

  v8f o[8];
  float mI[8], lI[8];
#pragma unroll
  for (int j = 0; j < 8; j++) {
    v8f z = {};
    o[j] = z;
    mI[j] = -1e30f;
    lI[j] = 0.f;
  }

  bf16_t* Pw = Ps + wid * (16 * 32);
  const int nt = qt * 4 + 4;  // key tiles needed for causal coverage
  const float scale = 0.08838834764831845f;  // 1/sqrt(128)

  stage_kv(Kv, VT, Ks, Vs, headBase, headBaseT, 0, tid);
  async_wait();
  __syncthreads();

  for (int t = 0; t < nt; ++t) {
    const int cur = t & 1;
    const bf16_t* Kc = Ks + cur * (32 * 128);
    const bf16_t* Vc = Vs + cur * (128 * 32);
    if (t + 1 < nt)
      stage_kv(Kv, VT, Ks + (1 - cur) * (32 * 128), Vs + (1 - cur) * (128 * 32),
               headBase, headBaseT, (t + 1) * 32, tid);

    // scores: 16 queries x 32 keys = two 16x16 C tiles
    const int kv0 = t * 32;
    v8f sa[2];
    {
      v8f z = {};
      sa[0] = z;
      sa[1] = z;
    }
#pragma unroll
    for (int j = 0; j < 2; j++)
#pragma unroll
      for (int kk = 0; kk < 4; kk++)
        sa[j] = wmma_bf16(qf[kk],
                          load_frag(Kc + (j * 16 + lr) * 128 + kk * 32, lh),
                          sa[j]);

    // online softmax per query row (row = v + 8*lh for this lane)
#pragma unroll
    for (int v = 0; v < 8; v++) {
      const int qg = q0 + wid * 16 + v + 8 * lh;
      float s0 = sa[0][v] * scale;
      float s1 = sa[1][v] * scale;
      s0 = ((kv0 + lr) <= qg) ? s0 : -1e30f;
      s1 = ((kv0 + 16 + lr) <= qg) ? s1 : -1e30f;
      float mx = fmaxf(s0, s1);
#pragma unroll
      for (int off = 1; off < 16; off <<= 1)
        mx = fmaxf(mx, __shfl_xor(mx, off, 32));
      const float mnew = fmaxf(mI[v], mx);
      const float p0 = __expf(s0 - mnew);
      const float p1 = __expf(s1 - mnew);
      const float alpha = __expf(mI[v] - mnew);
      float rs = p0 + p1;
#pragma unroll
      for (int off = 1; off < 16; off <<= 1) rs += __shfl_xor(rs, off, 32);
      lI[v] = lI[v] * alpha + rs;
      mI[v] = mnew;
#pragma unroll
      for (int jo = 0; jo < 8; jo++) o[jo][v] *= alpha;
      Pw[(v + 8 * lh) * 32 + lr] = (bf16_t)p0;
      Pw[(v + 8 * lh) * 32 + 16 + lr] = (bf16_t)p1;
    }

    // P (16x32) as A-fragment, then O += P * V
    const v16bf pf = load_frag(Pw + lr * 32, lh);
#pragma unroll
    for (int jo = 0; jo < 8; jo++)
      o[jo] = wmma_bf16(pf, load_frag(Vc + (jo * 16 + lr) * 32, lh), o[jo]);

    async_wait();
    __syncthreads();
  }

  // normalize and write to [b, s, h*dh] (row-major bf16 for the out-proj GEMM)
#pragma unroll
  for (int jo = 0; jo < 8; jo++)
#pragma unroll
    for (int v = 0; v < 8; v++) {
      const int qg = q0 + wid * 16 + v + 8 * lh;
      const int dh = jo * 16 + lr;
      const float val = o[jo][v] / lI[v];
      AO[((size_t)b * SL + qg) * DM + h * DH + dh] = (bf16_t)val;
    }
}

// ---------------------------------------------------------------------------
// Launch
// ---------------------------------------------------------------------------
extern "C" void kernel_launch(void* const* d_in, const int* in_sizes, int n_in,
                              void* d_out, int out_size, void* d_ws,
                              size_t ws_size, hipStream_t stream) {
  (void)in_sizes; (void)n_in; (void)out_size; (void)ws_size;
  const float* x = (const float*)d_in[0];
  const float* kqvw = (const float*)d_in[1];
  const float* outw = (const float*)d_in[2];
  float* out = (float*)d_out;

  char* ws = (char*)d_ws;
  const size_t SZ_X = (size_t)BSM * DM * 2;  // 32 MiB
  const size_t SZ_W3 = (size_t)3 * DM * DM * 2;
  const size_t SZ_W = (size_t)DM * DM * 2;
  bf16_t* xb = (bf16_t*)ws;  ws += SZ_X;
  bf16_t* w3b = (bf16_t*)ws; ws += SZ_W3;
  bf16_t* wob = (bf16_t*)ws; ws += SZ_W;
  bf16_t* qb = (bf16_t*)ws;  ws += SZ_X;
  bf16_t* kb = (bf16_t*)ws;  ws += SZ_X;
  bf16_t* vtb = (bf16_t*)ws; ws += SZ_X;
  bf16_t* aob = xb;  // reuse x's bf16 buffer: x is dead after the QKV GEMM

  cvt_kernel<<<(BSM * DM / 4 + 255) / 256, 256, 0, stream>>>(x, xb,
                                                             BSM * DM / 4);
  cvt_kernel<<<(3 * DM * DM / 4 + 255) / 256, 256, 0, stream>>>(
      kqvw, w3b, 3 * DM * DM / 4);
  cvt_kernel<<<(DM * DM / 4 + 255) / 256, 256, 0, stream>>>(outw, wob,
                                                            DM * DM / 4);

  qkv_gemm_kernel<<<dim3(DM / 128, BSM / 128, 3), 256, 0, stream>>>(
      xb, w3b, qb, kb, vtb);
  rope_kernel<<<(NB * NH * SL * 64) / 256, 256, 0, stream>>>(qb, kb);
  flash_kernel<<<dim3(SL / 128, NH, NB), 256, 0, stream>>>(qb, kb, vtb, aob);
  out_gemm_kernel<<<dim3(DM / 128, BSM / 128), 256, 0, stream>>>(aob, wob, out);
}
